// BidirRecurrentModel_1683627180774
// MI455X (gfx1250) — matrix operand order repaired and checked by reading, not compiled, not run on hardware
//
#include <hip/hip_runtime.h>

// ---------------------------------------------------------------------------
// BidirLSTM (2 layers, shared weights between directions) + FC, MI455X/gfx1250
//   fwd: full 256-step recurrence per layer (one kernel launch per step)
//   bwd: bwd[0] depends only on x[:,T-1,:] with zero init state -> 2 steps
//   precision: bf16 WMMA (v_wmma_f32_16x16x32_bf16) with f32 accumulation
//   this round: depth-4 modulo software pipeline held in place with
//   __builtin_amdgcn_sched_barrier(0) per stage + __launch_bounds__(256,2)
//   so the scheduler/RA cannot sink loads back onto the WMMAs; every WMMA
//   waits only for loads issued 4 iterations earlier (30 newer loads in
//   flight -> partial s_wait_loadcnt, L2 latency off the critical path).
// ---------------------------------------------------------------------------

typedef __attribute__((ext_vector_type(16))) __bf16        v16bf;
typedef __attribute__((ext_vector_type(8)))  float         v8f;
typedef __attribute__((ext_vector_type(4))) unsigned int   u32x4;

struct B32B { u32x4 lo, hi; };   // 32 bytes == one v16bf fragment

#define BB 64     // batch
#define TT 256    // timesteps
#define HH 1024   // hidden (== IN == OUT)
#define KK 1024   // contraction dim of every LSTM projection

#if defined(__has_builtin)
#if __has_builtin(__builtin_amdgcn_sched_barrier)
#define SCHED_FENCE() __builtin_amdgcn_sched_barrier(0)
#endif
#endif
#ifndef SCHED_FENCE
#define SCHED_FENCE() asm volatile("" ::: "memory")
#endif

// round-to-nearest-even f32 -> bf16
__device__ __forceinline__ __bf16 f2bf(float f) {
  unsigned u = __builtin_bit_cast(unsigned, f);
  unsigned r = (u + 0x7FFFu + ((u >> 16) & 1u)) >> 16;
  unsigned short s = (unsigned short)r;
  return __builtin_bit_cast(__bf16, s);
}

__device__ __forceinline__ float sigm(float x) {
  x = fminf(fmaxf(x, -30.f), 30.f);
  return 1.f / (1.f + __expf(-x));
}
__device__ __forceinline__ float tanh_f(float x) {
  x = fminf(fmaxf(x, -15.f), 15.f);
  float e = __expf(-2.f * x);
  return (1.f - e) / (1.f + e);
}

// ---------------------------------------------------------------------------
// Software-pipelined 4-accumulator GEMM slice.
//   KT   : number of K-iterations (compile-time, fully unrolled)
//   GSTR : stride between the 4 B strips, in bf16 elements (folds into the
//          VMEM 24-bit immediate: g*GSTR*2 bytes, max 6 MB < 8.38 MB)
//   PF   : prefetch depth (stages in flight)
// Per iteration: 4 WMMAs on stage kt%PF, then 10 loads refilling that stage
// for kt+PF, then a sched_barrier pinning the order.
// ---------------------------------------------------------------------------
template <int KT, long GSTR, int PF>
__device__ __forceinline__ void gemm4_pipe(
    v8f acc[4], const __bf16* __restrict__ A, const __bf16* __restrict__ Bp,
    int tb0, int arow, int lid, int ldk) {
  const __bf16* pa  = A  + (size_t)arow * ldk + ((lid >> 4) << 3);
  const __bf16* pb0 = Bp + (size_t)lid * 16 + (size_t)tb0 * 512;

  auto ldA = [&](int kt) {
    B32B t;
    t.lo = *(const u32x4*)(pa + (kt << 5));        // K kt*32 + half*8 .. +7
    t.hi = *(const u32x4*)(pa + (kt << 5) + 16);   // K kt*32+16+half*8 .. +7
    return __builtin_bit_cast(v16bf, t);
  };
  auto ldB = [&](long g, long kt) {
    const __bf16* p = pb0 + g * GSTR + (kt << 9);
    B32B t;
    t.lo = *(const u32x4*)(p);
    t.hi = *(const u32x4*)(p + 8);
    return __builtin_bit_cast(v16bf, t);
  };

  v16bf a[PF];
  v16bf b[PF][4];
#pragma unroll
  for (int s = 0; s < PF; ++s) {           // prologue: fill PF stages
    const int k = (s < KT) ? s : KT - 1;
    a[s] = ldA(k);
#pragma unroll
    for (int g = 0; g < 4; ++g) b[s][g] = ldB(g, k);
  }
  SCHED_FENCE();

#pragma unroll
  for (int kt = 0; kt < KT; ++kt) {
    const int s = kt % PF;
    acc[0] = __builtin_amdgcn_wmma_f32_16x16x32_bf16(
        false, a[s], false, b[s][0], (short)0, acc[0], false, false);
    acc[1] = __builtin_amdgcn_wmma_f32_16x16x32_bf16(
        false, a[s], false, b[s][1], (short)0, acc[1], false, false);
    acc[2] = __builtin_amdgcn_wmma_f32_16x16x32_bf16(
        false, a[s], false, b[s][2], (short)0, acc[2], false, false);
    acc[3] = __builtin_amdgcn_wmma_f32_16x16x32_bf16(
        false, a[s], false, b[s][3], (short)0, acc[3], false, false);

    // refill this stage for iteration kt+PF (clamped -> redundant in-bounds
    // loads on the last PF iterations; keeps the body uniform)
    const int kn = (kt + PF < KT) ? kt + PF : KT - 1;
    a[s] = ldA(kn);
#pragma unroll
    for (int g = 0; g < 4; ++g) b[s][g] = ldB(g, kn);
    SCHED_FENCE();
  }
}

// ---------------------------------------------------------------------------
// Pack f32 weight W[N][K] (row-major, gates = in @ W^T) into bf16 B-fragments.
// B element e (h=lane>>4): n=n0+(lane&15), k=k0+8h+e (e<8), k0+16+8h+(e-8) (e>=8)
// ---------------------------------------------------------------------------
__global__ void pack_b(const float* __restrict__ W, __bf16* __restrict__ out,
                       int N, int K) {
  int t = blockIdx.x * blockDim.x + threadIdx.x;
  int total = (N >> 4) * (K >> 5) * 32;
  if (t >= total) return;
  int lane   = t & 31;
  int tile   = t >> 5;
  int ktiles = K >> 5;
  int kt = tile % ktiles;
  int nt = tile / ktiles;
  int n  = (nt << 4) + (lane & 15);
  int kb = (kt << 5) + ((lane >> 4) << 3);
  __bf16* o = out + (size_t)t * 16;
  const float* w = W + (size_t)n * K + kb;
#pragma unroll
  for (int e = 0; e < 16; ++e) o[e] = f2bf(w[e < 8 ? e : e + 8]);
}

// x [B][T][IN] f32  ->  xt [T][B][IN] bf16
__global__ void xpose_cast(const float* __restrict__ x, __bf16* __restrict__ xt) {
  long i = (long)blockIdx.x * blockDim.x + threadIdx.x;   // T*B*IN elements
  if (i >= (long)TT * BB * KK) return;
  int  k  = (int)(i & (KK - 1));
  long rb = i >> 10;
  int  b  = (int)(rb & (BB - 1));
  int  t  = (int)(rb >> 6);
  xt[i] = f2bf(x[((long)b * TT + t) * KK + k]);
}

// ---------------------------------------------------------------------------
// One LSTM cell step for all B=64 rows.
// grid = 32 WGs (each owns 32 h-columns), block = 256 threads = 8 waves.
// Wave (mt = wid&3, hc = wid>>2) owns rows [mt*16,+16) x h-cols [base+hc*16,+16)
// and accumulates gates i,f,g,o for that tile in 4 f32 accumulators, so the
// nonlinearity needs no cross-wave exchange.
// ---------------------------------------------------------------------------
__global__ __launch_bounds__(256, 2)
void lstm_step(const __bf16* __restrict__ in_t,    // [64][1024] bf16
               const __bf16* __restrict__ h_prev,  // [64][1024] bf16 (unused if first)
               const __bf16* __restrict__ Wihp,    // packed [4H][K]
               const __bf16* __restrict__ Whhp,    // packed [4H][H]
               const float*  __restrict__ b_ih,
               const float*  __restrict__ b_hh,
               float*        __restrict__ c_state, // [64][1024] f32 in/out
               __bf16*       __restrict__ h_out,   // [64][1024] bf16
               int first) {
  const int lid = threadIdx.x & 31;
  const int wid = threadIdx.x >> 5;
  const int mt  = wid & 3;
  const int hc  = wid >> 2;
  const int m0  = mt << 4;
  const int nh0 = blockIdx.x * 32 + hc * 16;   // h-column base of this wave
  constexpr int  KT   = KK >> 5;               // 32 k-iterations per projection
  constexpr long GSTR = (long)(HH >> 4) * KT * 512;  // 1,048,576 elems = 2 MB
  constexpr int  PF   = 4;                     // prefetch depth

  v8f acc[4];
#pragma unroll
  for (int g = 0; g < 4; ++g) acc[g] = {};

  const int tb0  = (nh0 >> 4) * KT;            // gate-0 tile base
  const int arow = m0 + (lid & 15);

  // input projection: gates += in_t @ W_ih^T
  gemm4_pipe<KT, GSTR, PF>(acc, in_t, Wihp, tb0, arow, lid, KK);
  // recurrent projection: gates += h_prev @ W_hh^T   (h0 == 0 at first step)
  if (!first)
    gemm4_pipe<KT, GSTR, PF>(acc, h_prev, Whhp, tb0, arow, lid, HH);

  const int col = nh0 + (lid & 15);
  float bias[4];
#pragma unroll
  for (int g = 0; g < 4; ++g) {
    int n = g * HH + col;
    bias[g] = b_ih[n] + b_hh[n];
  }

  const int rbase = m0 + ((lid >> 4) << 3);    // C/D layout: elem r -> row rbase+r
#pragma unroll
  for (int r = 0; r < 8; ++r) {
    int row  = rbase + r;
    long idx = (long)row * HH + col;
    float gi = sigm(acc[0][r] + bias[0]);
    float gf = sigm(acc[1][r] + bias[1]);
    float gg = tanh_f(acc[2][r] + bias[2]);
    float go = sigm(acc[3][r] + bias[3]);
    float cp = first ? 0.f : c_state[idx];
    float c2 = gf * cp + gi * gg;
    float h2 = go * tanh_f(c2);
    c_state[idx] = c2;
    h_out[idx]   = f2bf(h2);
  }
}

// hcat[b][0:H] = fwd_last, hcat[b][H:2H] = bwd_first
__global__ void concat_h(const __bf16* __restrict__ f,
                         const __bf16* __restrict__ bw,
                         __bf16* __restrict__ hcat) {
  int i = blockIdx.x * blockDim.x + threadIdx.x;   // 64*2048
  if (i >= BB * 2 * HH) return;
  int j = i & (2 * HH - 1);
  int b = i >> 11;
  hcat[i] = (j < HH) ? f[b * HH + j] : bw[b * HH + (j - HH)];
}

// out[64][1024] = hcat[64][2048] @ fcW^T + fc_b   (bf16 WMMA, f32 out)
// adjacent output n-tiles are KT tiles apart in the packed buffer:
// GSTR = KT*512 = 32768 elements = 64 KB.
__global__ __launch_bounds__(256, 2)
void fc_gemm(const __bf16* __restrict__ A,    // [64][2048] bf16
             const __bf16* __restrict__ Bp,   // packed N=1024, K=2048
             const float*  __restrict__ bias,
             float*        __restrict__ out) {
  const int lid = threadIdx.x & 31;
  const int wid = threadIdx.x >> 5;
  const int mt  = wid & 3;
  const int ng  = wid >> 2;
  const int m0  = mt << 4;
  const int nbase = blockIdx.x * 128 + ng * 64;
  constexpr int  K    = 2 * HH;
  constexpr int  N    = HH;
  constexpr int  KT   = K >> 5;                // 64
  constexpr long GSTR = (long)KT * 512;        // 32,768 elements
  constexpr int  PF   = 4;

  v8f acc[4];
#pragma unroll
  for (int j = 0; j < 4; ++j) acc[j] = {};

  const int tb0  = (nbase >> 4) * KT;
  const int arow = m0 + (lid & 15);
  gemm4_pipe<KT, GSTR, PF>(acc, A, Bp, tb0, arow, lid, K);

  const int rbase = m0 + ((lid >> 4) << 3);
#pragma unroll
  for (int j = 0; j < 4; ++j) {
    int col  = nbase + j * 16 + (lid & 15);
    float bs = bias[col];
#pragma unroll
    for (int r = 0; r < 8; ++r)
      out[(long)(rbase + r) * N + col] = acc[j][r] + bs;
  }
}

// ---------------------------------------------------------------------------
extern "C" void kernel_launch(void* const* d_in, const int* in_sizes, int n_in,
                              void* d_out, int out_size, void* d_ws, size_t ws_size,
                              hipStream_t stream) {
  (void)in_sizes; (void)n_in; (void)out_size; (void)ws_size;
  const float* x    = (const float*)d_in[0];
  const float* Wih0 = (const float*)d_in[1];
  const float* Whh0 = (const float*)d_in[2];
  const float* bih0 = (const float*)d_in[3];
  const float* bhh0 = (const float*)d_in[4];
  const float* Wih1 = (const float*)d_in[5];
  const float* Whh1 = (const float*)d_in[6];
  const float* bih1 = (const float*)d_in[7];
  const float* bhh1 = (const float*)d_in[8];
  const float* fcW  = (const float*)d_in[9];
  const float* fcb  = (const float*)d_in[10];
  float* out = (float*)d_out;

  // ---- workspace carve-up (all offsets 256B aligned) ----
  char* ws = (char*)d_ws;
  size_t off = 0;
  auto alloc = [&](size_t bytes) -> char* {
    char* p = ws + off;
    off = (off + bytes + 255) & ~(size_t)255;
    return p;
  };
  __bf16* xt    = (__bf16*)alloc((size_t)TT * BB * KK * 2);  // 32 MB
  __bf16* Wih0p = (__bf16*)alloc((size_t)4 * HH * KK * 2);   // 8 MB each
  __bf16* Whh0p = (__bf16*)alloc((size_t)4 * HH * HH * 2);
  __bf16* Wih1p = (__bf16*)alloc((size_t)4 * HH * HH * 2);
  __bf16* Whh1p = (__bf16*)alloc((size_t)4 * HH * HH * 2);
  __bf16* fcWp  = (__bf16*)alloc((size_t)HH * 2 * HH * 2);   // 4 MB
  __bf16* h0seq = (__bf16*)alloc((size_t)TT * BB * HH * 2);  // 32 MB
  float*  c0    = (float*) alloc((size_t)BB * HH * 4);
  float*  c1    = (float*) alloc((size_t)BB * HH * 4);
  __bf16* h1pp  = (__bf16*)alloc((size_t)2 * BB * HH * 2);   // ping-pong
  __bf16* h0b   = (__bf16*)alloc((size_t)BB * HH * 2);
  __bf16* h1b   = (__bf16*)alloc((size_t)BB * HH * 2);
  __bf16* hcat  = (__bf16*)alloc((size_t)BB * 2 * HH * 2);

  // ---- one-time prep: cast/transpose x, pack all weights for WMMA-B ----
  xpose_cast<<<(TT * BB * KK) / 256, 256, 0, stream>>>(x, xt);
  pack_b<<<1024, 256, 0, stream>>>(Wih0, Wih0p, 4 * HH, KK);
  pack_b<<<1024, 256, 0, stream>>>(Whh0, Whh0p, 4 * HH, HH);
  pack_b<<<1024, 256, 0, stream>>>(Wih1, Wih1p, 4 * HH, HH);
  pack_b<<< 512, 256, 0, stream>>>(fcW,  fcWp,  HH, 2 * HH);
  pack_b<<<1024, 256, 0, stream>>>(Whh1, Whh1p, 4 * HH, HH);

  // ---- forward, layer 0: write whole output sequence ----
  for (int t = 0; t < TT; ++t) {
    const __bf16* in = xt    + (size_t)t * BB * KK;
    const __bf16* hp = h0seq + (size_t)(t > 0 ? t - 1 : 0) * BB * HH;
    __bf16*       ho = h0seq + (size_t)t * BB * HH;
    lstm_step<<<32, 256, 0, stream>>>(in, hp, Wih0p, Whh0p, bih0, bhh0,
                                      c0, ho, t == 0);
  }
  // ---- forward, layer 1: ping-pong h ----
  for (int t = 0; t < TT; ++t) {
    const __bf16* in = h0seq + (size_t)t * BB * HH;
    const __bf16* hp = h1pp + (size_t)((t + 1) & 1) * BB * HH;
    __bf16*       ho = h1pp + (size_t)(t & 1) * BB * HH;
    lstm_step<<<32, 256, 0, stream>>>(in, hp, Wih1p, Whh1p, bih1, bhh1,
                                      c1, ho, t == 0);
  }
  const __bf16* fwd_last = h1pp + (size_t)((TT - 1) & 1) * BB * HH;

  // ---- backward direction: bwd[0] needs exactly one step per layer ----
  lstm_step<<<32, 256, 0, stream>>>(xt + (size_t)(TT - 1) * BB * KK, h0b,
                                    Wih0p, Whh0p, bih0, bhh0, c0, h0b, 1);
  lstm_step<<<32, 256, 0, stream>>>(h0b, h1b, Wih1p, Whh1p, bih1, bhh1,
                                    c1, h1b, 1);

  // ---- concat + FC head ----
  concat_h<<<(BB * 2 * HH) / 256, 256, 0, stream>>>(fwd_last, h1b, hcat);
  fc_gemm<<<HH / 128, 256, 0, stream>>>(hcat, fcWp, fcb, out);
}
